// RelationAwareTreeLSTMCell_72516227825749
// MI455X (gfx1250) — matrix operand order repaired and checked by reading, not compiled, not run on hardware
//
#include <hip/hip_runtime.h>
#include <hip/hip_bf16.h>
#include <math.h>

// ---------------------------------------------------------------------------
// RelationAwareTreeLSTMCell for MI455X (gfx1250, wave32, WMMA)
// B=4096 N=32 D_IN=256 H=256 R=3
// ---------------------------------------------------------------------------

#define Bsz   4096
#define Nch   32
#define Hd    256
#define Din   256
#define Rn    3
#define CDIM  512           // D_IN + H
#define G3    768           // 3 gates * H
#define VIRT_DECAY 0.7f

typedef __attribute__((ext_vector_type(16))) __bf16       v16bf;
typedef __attribute__((ext_vector_type(8)))  float        v8f;
typedef __attribute__((ext_vector_type(8)))  unsigned int v8u;

// ---- helpers ---------------------------------------------------------------

__device__ __forceinline__ unsigned short f2bf(float x) {
  unsigned int u = __builtin_bit_cast(unsigned int, x);
  unsigned int r = u + 0x7FFFu + ((u >> 16) & 1u);   // round-to-nearest-even
  return (unsigned short)(r >> 16);
}

__device__ __forceinline__ v16bf pack_frag(uint4 lo, uint4 hi) {
  v8u u;
  u[0] = lo.x; u[1] = lo.y; u[2] = lo.z; u[3] = lo.w;
  u[4] = hi.x; u[5] = hi.y; u[6] = hi.z; u[7] = hi.w;
  return __builtin_bit_cast(v16bf, u);
}

// A-matrix 16x32 bf16 fragment: lane m = lane&15, hi half selects
// K range {16h..16h+15} -> two contiguous 16B LDS loads.
__device__ __forceinline__ v16bf load_a(const unsigned short* p0, int row,
                                        int stride, int k0, int lane) {
  int h8 = (lane >> 4) << 4;
  const unsigned short* p = p0 + row * stride + k0 + h8;
  uint4 lo = *(const uint4*)p;
  uint4 hi = *(const uint4*)(p + 8);
  return pack_frag(lo, hi);
}

// B fragment pre-packed in memory: 8 contiguous dwords per lane.
__device__ __forceinline__ v16bf load_b(const unsigned int* pack, int idx8) {
  const uint4* p = (const uint4*)(pack + ((long)idx8 << 3));
  return pack_frag(p[0], p[1]);
}

__device__ __forceinline__ float wave_sum(float v) {
#pragma unroll
  for (int m = 16; m > 0; m >>= 1) v += __shfl_xor(v, m, 32);
  return v;
}
__device__ __forceinline__ float wave_max(float v) {
#pragma unroll
  for (int m = 16; m > 0; m >>= 1) v = fmaxf(v, __shfl_xor(v, m, 32));
  return v;
}

// ---- weight packing: B[k][n] fragment order, per-lane contiguous ------------
// B element (k,n): lane = (n&15) + 16*hi, dword j holds K = k0 + 16*hi + 2j (+1)

// W_f (R,H,H): B[k][g] = W_f[r][g][k].  dims: r(3) kt(8) nt(16) lane(32) j(8)
__global__ void pack_wf_kernel(const float* __restrict__ Wf,
                               unsigned int* __restrict__ out) {
  int idx = blockIdx.x * 256 + threadIdx.x;         // < 98304
  int j    = idx & 7;
  int lane = (idx >> 3) & 31;
  int nt   = (idx >> 8) & 15;
  int kt   = (idx >> 12) & 7;
  int r    = idx >> 15;
  int hi   = lane >> 4;
  int g    = nt * 16 + (lane & 15);
  int k    = kt * 32 + 16 * hi + 2 * j;
  const float* row = Wf + ((long)r * Hd + g) * Hd;
  unsigned int d = (unsigned int)f2bf(row[k]) |
                   ((unsigned int)f2bf(row[k + 1]) << 16);
  out[idx] = d;
}

// Gates: Wg = [W_i; W_o; W_u] (768 x 512), B[k][n] = Wg[n][k].
// dims: kt(16) nt(48) lane(32) j(8) -> 196608 dwords
__global__ void pack_wg_kernel(const float* __restrict__ Wi,
                               const float* __restrict__ Wo,
                               const float* __restrict__ Wu,
                               unsigned int* __restrict__ out) {
  int idx = blockIdx.x * 256 + threadIdx.x;         // < 196608
  int kt   = idx / 12288;
  int rem  = idx - kt * 12288;
  int nt   = rem >> 8;
  int rem2 = rem & 255;
  int lane = rem2 >> 3;
  int j    = rem2 & 7;
  int hi   = lane >> 4;
  int n    = nt * 16 + (lane & 15);
  int k    = kt * 32 + 16 * hi + 2 * j;
  const float* row = (n < 256) ? (Wi + (long)n * CDIM)
                   : (n < 512) ? (Wo + (long)(n - 256) * CDIM)
                               : (Wu + (long)(n - 512) * CDIM);
  out[idx] = (unsigned int)f2bf(row[k]) |
             ((unsigned int)f2bf(row[k + 1]) << 16);
}

// ---- main fused kernel: one workgroup (256 thr / 8 waves) per batch row ----
__global__ __launch_bounds__(256)
void treelstm_main_kernel(const float* __restrict__ input_vec,
                          const float* __restrict__ child_h,
                          const float* __restrict__ child_c,
                          const int* __restrict__ relation_ids,
                          const unsigned char* __restrict__ virtual_mask, // jax bool = 1B
                          const float* __restrict__ rel_emb,
                          const float* __restrict__ w_att,
                          const float* __restrict__ b_att,
                          const float* __restrict__ b_f,
                          unsigned short* __restrict__ combined,   // (B,512) bf16
                          const unsigned int* __restrict__ BpackF,
                          float* __restrict__ csum_out) {          // (B,256)
  __shared__ __align__(16) float          chs[Nch * Hd];      // decayed child_h f32
  __shared__ __align__(16) unsigned short chb[Nch * Hd];      // bf16 copy (GEMM A)
  __shared__ float s_scores[Nch];
  __shared__ float s_attn[Nch];
  __shared__ int   s_rel[Nch];
  __shared__ float s_decay[Nch];
  __shared__ float s_bf[Rn * Hd];
  __shared__ float s_csum[Hd];

  const int b    = blockIdx.x;
  const int t    = threadIdx.x;
  const int w    = t >> 5;
  const int lane = t & 31;
  const int hi8  = (lane >> 4) << 3;
  const int nlan = lane & 15;

  // phase 0: metadata + zero accumulators
  if (t < Nch) {
    s_rel[t]   = relation_ids[b * Nch + t];
    s_decay[t] = virtual_mask[b * Nch + t] ? VIRT_DECAY : 1.0f;
  }
  if (t < Hd) s_csum[t] = 0.0f;
#pragma unroll
  for (int r = 0; r < Rn; ++r) s_bf[r * Hd + t] = b_f[r * Hd + t];
  // input half of "combined" in bf16
  combined[(long)b * CDIM + t] = f2bf(input_vec[(long)b * Din + t]);
  __syncthreads();

  // phase 1: decayed ch into LDS (f32 + bf16)
  for (int n = 0; n < Nch; ++n) {
    float v = child_h[((long)b * Nch + n) * Hd + t] * s_decay[n];
    chs[n * Hd + t] = v;
    chb[n * Hd + t] = f2bf(v);
  }
  __syncthreads();

  // phase 2: attention scores (wave per 4 children)
#pragma unroll
  for (int i = 0; i < 4; ++i) {
    int n = w * 4 + i;
    int r = s_rel[n];
    float p = 0.0f;
#pragma unroll
    for (int jj = 0; jj < 8; ++jj) {
      int g = lane + 32 * jj;
      p += (rel_emb[r * Hd + g] + chs[n * Hd + g]) * w_att[g];
    }
    p = wave_sum(p);
    if (lane == 0) s_scores[n] = p + b_att[0];
  }
  __syncthreads();

  // phase 3: softmax over 32 children (wave 0)
  if (t < Nch) {
    float s  = s_scores[t];
    float mx = wave_max(s);
    float e  = expf(s - mx);
    float sm = wave_sum(e);
    s_attn[t] = e / sm;
  }
  __syncthreads();

  // phase 4: child_h_sum -> combined[:,256:512] bf16
  {
    float acc = 0.0f;
#pragma unroll
    for (int n = 0; n < Nch; ++n) acc += chs[n * Hd + t] * s_attn[n];
    combined[(long)b * CDIM + Hd + t] = f2bf(acc);
  }

  // phase 5: WMMA GEMM  f_all = ch @ W_f[r]^T for r=0..2, per-row select
  // wave w owns nt in {w, w+8}, mt in {0,1}  (M=32, N=256, K=256)
  v8f fsel[2][2];
#pragma unroll
  for (int mt = 0; mt < 2; ++mt)
#pragma unroll
    for (int nti = 0; nti < 2; ++nti) {
      int ntv = w + nti * 8;
      int g   = ntv * 16 + nlan;
#pragma unroll
      for (int v = 0; v < 8; ++v) {
        int row = mt * 16 + v + hi8;
        fsel[mt][nti][v] = s_bf[s_rel[row] * Hd + g];   // bias b_f[rel,g]
      }
    }

#pragma unroll
  for (int r = 0; r < Rn; ++r) {
    v8f acc[2][2];
#pragma unroll
    for (int mt = 0; mt < 2; ++mt)
#pragma unroll
      for (int nti = 0; nti < 2; ++nti) acc[mt][nti] = (v8f)0.0f;

    for (int kt = 0; kt < 8; ++kt) {
      v16bf a0 = load_a(chb, (lane & 15),      Hd, kt * 32, lane);
      v16bf a1 = load_a(chb, 16 + (lane & 15), Hd, kt * 32, lane);
      v16bf b0 = load_b(BpackF, ((r * 8 + kt) * 16 + w)     * 32 + lane);
      v16bf b1 = load_b(BpackF, ((r * 8 + kt) * 16 + w + 8) * 32 + lane);
      acc[0][0] = __builtin_amdgcn_wmma_f32_16x16x32_bf16(false, a0, false, b0,
                    (short)0, acc[0][0], false, false);
      acc[1][0] = __builtin_amdgcn_wmma_f32_16x16x32_bf16(false, a1, false, b0,
                    (short)0, acc[1][0], false, false);
      acc[0][1] = __builtin_amdgcn_wmma_f32_16x16x32_bf16(false, a0, false, b1,
                    (short)0, acc[0][1], false, false);
      acc[1][1] = __builtin_amdgcn_wmma_f32_16x16x32_bf16(false, a1, false, b1,
                    (short)0, acc[1][1], false, false);
    }
    // per-row relation select
#pragma unroll
    for (int mt = 0; mt < 2; ++mt)
#pragma unroll
      for (int nti = 0; nti < 2; ++nti)
#pragma unroll
        for (int v = 0; v < 8; ++v) {
          int row = mt * 16 + v + hi8;
          if (s_rel[row] == r) fsel[mt][nti][v] += acc[mt][nti][v];
        }
  }

  // epilogue: child_c_sum[g] = sum_n f[n,g] * (child_c*decay)[n,g]
#pragma unroll
  for (int mt = 0; mt < 2; ++mt)
#pragma unroll
    for (int nti = 0; nti < 2; ++nti) {
      int ntv = w + nti * 8;
      int g   = ntv * 16 + nlan;
      float s = 0.0f;
#pragma unroll
      for (int v = 0; v < 8; ++v) {
        int row = mt * 16 + v + hi8;
        float cc = child_c[((long)b * Nch + row) * Hd + g] * s_decay[row];
        s += fsel[mt][nti][v] * cc;
      }
      atomicAdd(&s_csum[g], s);
    }
  __syncthreads();
  if (t < Hd) csum_out[(long)b * Hd + t] = s_csum[t];
}

// ---- gates GEMM: (4096 x 512) @ (512 x 768) bf16 WMMA ----------------------
__global__ __launch_bounds__(256)
void gates_gemm_kernel(const unsigned int* __restrict__ combined_u32, // bf16 pairs
                       const unsigned int* __restrict__ BpackG,
                       float* __restrict__ gates) {                   // (B,768)
  __shared__ __align__(16) unsigned int s_a[Nch * (CDIM / 2)];  // 32 rows x 512 bf16

  const int t    = threadIdx.x;
  const int w    = t >> 5;
  const int lane = t & 31;
  const int hi8  = (lane >> 4) << 3;
  const int nlan = lane & 15;
  const int mblk = blockIdx.x & 127;        // 128 M-blocks of 32 rows
  const int nblk = blockIdx.x >> 7;         // 6 N-blocks of 128 cols
  const int ntg  = nblk * 8 + w;            // global 16-col subtile (0..47)

  // cooperative A-tile load (32 x 512 bf16 = 8192 dwords)
#pragma unroll
  for (int i = 0; i < 32; ++i) {
    int u   = i * 256 + t;
    int row = u >> 8;
    int col = u & 255;
    s_a[row * 256 + col] = combined_u32[((long)mblk * 32 + row) * 256 + col];
  }
  __syncthreads();

  const unsigned short* a_lds = (const unsigned short*)s_a;
  v8f acc0 = (v8f)0.0f, acc1 = (v8f)0.0f;
  for (int kt = 0; kt < 16; ++kt) {
    v16bf a0 = load_a(a_lds, (lane & 15),      CDIM, kt * 32, lane);
    v16bf a1 = load_a(a_lds, 16 + (lane & 15), CDIM, kt * 32, lane);
    v16bf bb = load_b(BpackG, (kt * 48 + ntg) * 32 + lane);
    acc0 = __builtin_amdgcn_wmma_f32_16x16x32_bf16(false, a0, false, bb,
             (short)0, acc0, false, false);
    acc1 = __builtin_amdgcn_wmma_f32_16x16x32_bf16(false, a1, false, bb,
             (short)0, acc1, false, false);
  }

  int col = ntg * 16 + nlan;
#pragma unroll
  for (int v = 0; v < 8; ++v) {
    int r0 = mblk * 32 + v + hi8;
    int r1 = r0 + 16;
    gates[(long)r0 * G3 + col] = acc0[v];
    gates[(long)r1 * G3 + col] = acc1[v];
  }
}

// ---- final elementwise: activations, c, h ----------------------------------
__global__ void finalize_kernel(const float* __restrict__ gates,
                                const float* __restrict__ csum,
                                const float* __restrict__ b_i,
                                const float* __restrict__ b_o,
                                const float* __restrict__ b_u,
                                float* __restrict__ out) {  // [h (B*H) | c (B*H)]
  long idx = (long)blockIdx.x * 256 + threadIdx.x;   // < B*H
  int  b   = (int)(idx >> 8);
  int  g   = (int)(idx & 255);
  const float* gr = gates + (long)b * G3;
  float iv = 1.0f / (1.0f + expf(-(gr[g]        + b_i[g])));
  float ov = 1.0f / (1.0f + expf(-(gr[256 + g]  + b_o[g])));
  float uv = tanhf(gr[512 + g] + b_u[g]);
  float c  = iv * uv + csum[idx];
  float h  = ov * tanhf(c);
  out[idx]                    = h;
  out[(long)Bsz * Hd + idx]   = c;
}

// ---------------------------------------------------------------------------
extern "C" void kernel_launch(void* const* d_in, const int* in_sizes, int n_in,
                              void* d_out, int out_size, void* d_ws, size_t ws_size,
                              hipStream_t stream) {
  const float*         input_vec = (const float*)d_in[0];
  const float*         child_h   = (const float*)d_in[1];
  const float*         child_c   = (const float*)d_in[2];
  const int*           rel_ids   = (const int*)d_in[3];
  const unsigned char* vmask     = (const unsigned char*)d_in[4]; // jax bool: 1 byte
  const float*         rel_emb   = (const float*)d_in[5];
  const float*         W_i       = (const float*)d_in[6];
  const float*         b_i       = (const float*)d_in[7];
  const float*         W_f       = (const float*)d_in[8];
  const float*         b_f       = (const float*)d_in[9];
  const float*         W_o       = (const float*)d_in[10];
  const float*         b_o       = (const float*)d_in[11];
  const float*         W_u       = (const float*)d_in[12];
  const float*         b_u       = (const float*)d_in[13];
  const float*         w_att     = (const float*)d_in[14];
  const float*         b_att     = (const float*)d_in[15];
  float*               out       = (float*)d_out;

  // workspace layout (all 16B aligned)
  char* ws = (char*)d_ws;
  unsigned short* combined = (unsigned short*)ws;                       //  4,194,304 B
  float*          csum     = (float*)(ws + 4194304);                    //  4,194,304 B
  float*          gates    = (float*)(ws + 8388608);                    // 12,582,912 B
  unsigned int*   BpackF   = (unsigned int*)(ws + 20971520);            //    393,216 B
  unsigned int*   BpackG   = (unsigned int*)(ws + 21364736);            //    786,432 B
  (void)ws_size; (void)in_sizes; (void)n_in; (void)out_size;

  pack_wf_kernel<<<384, 256, 0, stream>>>(W_f, BpackF);
  pack_wg_kernel<<<768, 256, 0, stream>>>(W_i, W_o, W_u, BpackG);

  treelstm_main_kernel<<<Bsz, 256, 0, stream>>>(
      input_vec, child_h, child_c, rel_ids, vmask, rel_emb, w_att, b_att, b_f,
      combined, BpackF, csum);

  gates_gemm_kernel<<<128 * 6, 256, 0, stream>>>(
      (const unsigned int*)combined, BpackG, gates);

  finalize_kernel<<<Bsz, 256, 0, stream>>>(gates, csum, b_i, b_o, b_u, out);
}
